// GGNN_34497177322109
// MI455X (gfx1250) — matrix operand order repaired and checked by reading, not compiled, not run on hardware
//
#include <hip/hip_runtime.h>
#include <hip/hip_bf16.h>

// GGNN on MI455X (gfx1250): wave32 WMMA f16 16x16x32 everywhere.
// One workgroup (256 thr = 8 waves) per graph; A + state live in LDS.

typedef __attribute__((ext_vector_type(16))) _Float16 v16h;
typedef __attribute__((ext_vector_type(8)))  _Float16 hv8;
typedef __attribute__((ext_vector_type(8)))  float    v8f;

#define NN 60   // nodes
#define DD 64   // state dim
#define EE 4    // edge types

// f16 weight workspace offsets (in halves), all 16B-aligned
#define OFF_WIN  0        // [g=64][k=256], k=e*64+d  -> W_in[e][g][d]
#define OFF_WOUT 16384    // same from W_out
#define OFF_WR   32768    // [g=64][k=192] direct copy of Wr
#define OFF_WZ   45056
#define OFF_WH   57344
#define OFF_W1   69632    // [g=64][k=96], k<77 valid
#define OFF_W2   75776
#define WS_HALVES 81920

union Frag { v16h v; hv8 h[2]; };

// A-operand (16x32 f16, MxK): lane m=lane&15; lane<16 holds K={kk..kk+7,kk+16..kk+23},
// lane>=16 holds K={kk+8..kk+15,kk+24..kk+31}. src row-major [m][ld].
__device__ inline v16h ldA(const _Float16* base, int ld, int m0, int kk, int lane) {
  int m   = m0 + (lane & 15);
  int sel = (lane >> 4) * 8;
  const _Float16* p = base + m * ld + kk + sel;
  Frag f;
  f.h[0] = *(const hv8*)(p);
  f.h[1] = *(const hv8*)(p + 16);
  return f.v;
}

// B-operand (32x16 f16, KxN): lane n=lane&15; lanes 0-15 hold K={kk..kk+7,kk+16..kk+23},
// lanes 16-31 hold K={kk+8..kk+15,kk+24..kk+31}. src stored row-major [n][ld] (k contiguous)
// so each lane reads two 16-byte chunks.
__device__ inline v16h ldB(const _Float16* base, int ld, int n0, int kk, int lane) {
  int n  = n0 + (lane & 15);
  int k0 = kk + (lane >> 4) * 8;
  const _Float16* p = base + n * ld + k0;
  Frag f;
  f.h[0] = *(const hv8*)(p);
  f.h[1] = *(const hv8*)(p + 16);
  return f.v;
}

__device__ inline v8f wmma16(v16h a, v16h b, v8f c) {
  return __builtin_amdgcn_wmma_f32_16x16x32_f16(false, a, false, b, (short)0, c, false, false);
}

__device__ inline float sigm(float x) { return 1.f / (1.f + __expf(-x)); }

// ---------------- weight conversion (runs once per launch) ----------------
__global__ void prep_weights(const float* __restrict__ W_in, const float* __restrict__ W_out,
                             const float* __restrict__ Wr,  const float* __restrict__ Wz,
                             const float* __restrict__ Wh,  const float* __restrict__ W1,
                             const float* __restrict__ W2,  _Float16* __restrict__ ws) {
  int i = blockIdx.x * blockDim.x + threadIdx.x;
  if (i >= WS_HALVES) return;
  float v = 0.f;
  if (i < OFF_WOUT)      { int j = i;            int g = j >> 8; int k = j & 255; v = W_in [(k >> 6) * 4096 + g * 64 + (k & 63)]; }
  else if (i < OFF_WR)   { int j = i - OFF_WOUT; int g = j >> 8; int k = j & 255; v = W_out[(k >> 6) * 4096 + g * 64 + (k & 63)]; }
  else if (i < OFF_WZ)   { v = Wr[i - OFF_WR]; }
  else if (i < OFF_WH)   { v = Wz[i - OFF_WZ]; }
  else if (i < OFF_W1)   { v = Wh[i - OFF_WH]; }
  else if (i < OFF_W2)   { int j = i - OFF_W1;   int g = j / 96; int k = j % 96;  v = (k < 77) ? W1[g * 77 + k] : 0.f; }
  else                   { int j = i - OFF_W2;   int g = j / 96; int k = j % 96;  v = (k < 77) ? W2[g * 77 + k] : 0.f; }
  ws[i] = (_Float16)v;
}

// ---------------- main GGNN kernel: one block per graph ----------------
__global__ void __launch_bounds__(256)
ggnn_kernel(const float* __restrict__ prop_state,
            const float* __restrict__ annotation,
            const float* __restrict__ Aadj,
            const float* __restrict__ br, const float* __restrict__ bz,
            const float* __restrict__ bh, const float* __restrict__ b1,
            const float* __restrict__ b2,
            const _Float16* __restrict__ ws,
            float* __restrict__ out) {
  extern __shared__ char smem[];
  _Float16* sA   = (_Float16*)smem;           // [8 blk][64 m][64 n] f16  (65536 B)
  _Float16* sT   = sA + 8 * 64 * 64;          // [64 m][256 k]  f16      (32768 B)
  float*    h32  = (float*)(sT + 64 * 256);   // [64][64] f32            (16384 B)
  _Float16* h16  = (_Float16*)(h32 + 64*64);  // [m][d] row-major        ( 8192 B)
  _Float16* h16T = h16  + 64 * 64;            // [d][m] (B-operand)      ( 8192 B)
  _Float16* aI   = h16T + 64 * 64;            // a_in  f16               ( 8192 B)
  _Float16* aO   = aI   + 64 * 64;            // a_out f16               ( 8192 B)
  _Float16* rh   = aO   + 64 * 64;            // r*h   f16               ( 8192 B)
  _Float16* ann  = rh   + 64 * 64;            // [64][32] annotation pad ( 4096 B)
  float*    oacc = (float*)(ann + 64 * 32);   // [64] column reduction   (  256 B)

  const int tid  = threadIdx.x;
  const int lane = tid & 31;
  const int w    = tid >> 5;
  const int b    = blockIdx.x;

  // --- load A once (reused across all 5 steps), f32 -> f16, zero padded ---
  for (int idx = tid; idx < 8 * 64 * 64; idx += 256) {
    int blk = idx >> 12, m = (idx >> 6) & 63, n = idx & 63;
    int io = blk >> 2, e = blk & 3;
    float v = 0.f;
    if (m < NN && n < NN)
      v = Aadj[(size_t)b * (NN * 2 * NN * EE) + m * (2 * NN * EE) + io * (NN * EE) + e * NN + n];
    sA[idx] = (_Float16)v;
  }
  // --- init state h (rows >= 60 start zero; A's zero columns and the readout mask ignore them) ---
  for (int idx = tid; idx < 64 * 64; idx += 256) {
    int m = idx >> 6, g = idx & 63;
    float v = (m < NN) ? prop_state[(size_t)b * NN * DD + m * DD + g] : 0.f;
    h32[idx] = v; h16[idx] = (_Float16)v; h16T[g * 64 + m] = (_Float16)v;
  }
  // --- annotation, padded 13 -> 32 ---
  for (int idx = tid; idx < 64 * 32; idx += 256) {
    int m = idx >> 5, k = idx & 31;
    float v = (m < NN && k < 13) ? annotation[(size_t)b * NN * 13 + m * 13 + k] : 0.f;
    ann[idx] = (_Float16)v;
  }
  if (tid < 64) oacc[tid] = 0.f;
  __syncthreads();

  const int gt  = w & 3;          // this wave's N-tile for 64-wide outputs
  const int mtb = (w >> 2) * 2;   // this wave's two M-tiles
  v8f zreg[2];

  for (int step = 0; step < 5; ++step) {
    // ==== messages: a_{in,out} = [A_e @ h] @ Wstack  (T in LDS between) ====
    for (int io = 0; io < 2; ++io) {
      // T[e][m][f] = A_e @ h : 64 tile jobs, 8 per wave
      for (int j = 0; j < 8; ++j) {
        int job = w + 8 * j;
        int e = job >> 4, mt = (job >> 2) & 3, ft = job & 3;
        v8f acc = {};
        for (int kk = 0; kk < 64; kk += 32) {
          v16h af = ldA(sA + (io * 4 + e) * 4096, 64, mt * 16, kk, lane);
          v16h bf = ldB(h16T, 64, ft * 16, kk, lane);   // B[k=node][n=f]
          acc = wmma16(af, bf, acc);
        }
        int col = e * 64 + ft * 16 + (lane & 15);
        int mr  = mt * 16 + (lane >> 4) * 8;
#pragma unroll
        for (int r = 0; r < 8; ++r) sT[(mr + r) * 256 + col] = (_Float16)acc[r];
      }
      __syncthreads();
      // a = T @ Wstack (K=256): 16 tiles, 2 per wave; B frags straight from L2
      const _Float16* wsStack = ws + (io == 0 ? OFF_WIN : OFF_WOUT);
      _Float16* aDst = (io == 0) ? aI : aO;
      for (int t = 0; t < 2; ++t) {
        int mt = mtb + t;
        v8f acc = {};
        for (int kk = 0; kk < 256; kk += 32) {
          v16h af = ldA(sT, 256, mt * 16, kk, lane);
          v16h bf = ldB(wsStack, 256, gt * 16, kk, lane);
          acc = wmma16(af, bf, acc);
        }
        int col = gt * 16 + (lane & 15);
        int mr  = mt * 16 + (lane >> 4) * 8;
#pragma unroll
        for (int r = 0; r < 8; ++r) aDst[(mr + r) * 64 + col] = (_Float16)acc[r];
      }
      __syncthreads();
    }
    // ==== gates r, z over K=192 = [a_in | a_out | h] ====
    for (int t = 0; t < 2; ++t) {
      int mt = mtb + t;
      v8f accR = {}, accZ = {};
      for (int seg = 0; seg < 3; ++seg) {
        const _Float16* aSrc = (seg == 0) ? aI : (seg == 1) ? aO : h16;
        for (int kk = 0; kk < 64; kk += 32) {
          v16h af  = ldA(aSrc, 64, mt * 16, kk, lane);
          v16h brf = ldB(ws + OFF_WR, 192, gt * 16, seg * 64 + kk, lane);
          v16h bzf = ldB(ws + OFF_WZ, 192, gt * 16, seg * 64 + kk, lane);
          accR = wmma16(af, brf, accR);
          accZ = wmma16(af, bzf, accZ);
        }
      }
      int col = gt * 16 + (lane & 15);
      int mr  = mt * 16 + (lane >> 4) * 8;
      float biR = br[col], biZ = bz[col];
#pragma unroll
      for (int r = 0; r < 8; ++r) {
        float rv = sigm(accR[r] + biR);
        zreg[t][r] = sigm(accZ[r] + biZ);
        rh[(mr + r) * 64 + col] = (_Float16)(rv * h32[(mr + r) * 64 + col]);
      }
    }
    __syncthreads();
    // ==== h_hat over K=192 = [a_in | a_out | r*h], then GRU update ====
    for (int t = 0; t < 2; ++t) {
      int mt = mtb + t;
      v8f acc = {};
      for (int seg = 0; seg < 3; ++seg) {
        const _Float16* aSrc = (seg == 0) ? aI : (seg == 1) ? aO : rh;
        for (int kk = 0; kk < 64; kk += 32) {
          v16h af = ldA(aSrc, 64, mt * 16, kk, lane);
          v16h bf = ldB(ws + OFF_WH, 192, gt * 16, seg * 64 + kk, lane);
          acc = wmma16(af, bf, acc);
        }
      }
      int col = gt * 16 + (lane & 15);
      int mr  = mt * 16 + (lane >> 4) * 8;
      float biH = bh[col];
#pragma unroll
      for (int r = 0; r < 8; ++r) {
        int m = mr + r;
        float hh = tanhf(acc[r] + biH);
        float z  = zreg[t][r];
        float hn = (1.f - z) * h32[m * 64 + col] + z * hh;
        h32[m * 64 + col]  = hn;
        h16[m * 64 + col]  = (_Float16)hn;
        h16T[col * 64 + m] = (_Float16)hn;
      }
    }
    __syncthreads();
  }

  // ==== readout: [h | annot](K=96 padded) @ {W1,W2}, gated sum over nodes ====
  for (int t = 0; t < 2; ++t) {
    int mt = mtb + t;
    v8f acc1 = {}, acc2 = {};
    for (int kk = 0; kk < 96; kk += 32) {
      v16h af  = (kk < 64) ? ldA(h16, 64, mt * 16, kk, lane)
                           : ldA(ann, 32, mt * 16, 0, lane);
      v16h b1f = ldB(ws + OFF_W1, 96, gt * 16, kk, lane);
      v16h b2f = ldB(ws + OFF_W2, 96, gt * 16, kk, lane);
      acc1 = wmma16(af, b1f, acc1);
      acc2 = wmma16(af, b2f, acc2);
    }
    int col = gt * 16 + (lane & 15);
    int mr  = mt * 16 + (lane >> 4) * 8;
    float bi1 = b1[col], bi2 = b2[col];
    float partial = 0.f;
#pragma unroll
    for (int r = 0; r < 8; ++r) {
      int m = mr + r;
      if (m < NN)
        partial += sigm(acc2[r] + bi2) * tanhf(acc1[r] + bi1);
    }
    atomicAdd(&oacc[col], partial);   // ds_add_f32 column reduction
  }
  __syncthreads();
  if (tid < 64) out[(size_t)b * 64 + tid] = oacc[tid];
}

extern "C" void kernel_launch(void* const* d_in, const int* in_sizes, int n_in,
                              void* d_out, int out_size, void* d_ws, size_t ws_size,
                              hipStream_t stream) {
  // dict order: prop_state, annotation, A, total_atoms, property_features,
  //             W_in, b_in, W_out, b_out, Wr, br, Wz, bz, Wh, bh, W1, b1, W2, b2
  const float* prop_state = (const float*)d_in[0];
  const float* annotation = (const float*)d_in[1];
  const float* Aadj       = (const float*)d_in[2];
  const float* W_in  = (const float*)d_in[5];
  const float* W_out = (const float*)d_in[7];
  const float* Wr = (const float*)d_in[9];
  const float* br = (const float*)d_in[10];
  const float* Wz = (const float*)d_in[11];
  const float* bz = (const float*)d_in[12];
  const float* Wh = (const float*)d_in[13];
  const float* bh = (const float*)d_in[14];
  const float* W1 = (const float*)d_in[15];
  const float* b1 = (const float*)d_in[16];
  const float* W2 = (const float*)d_in[17];
  const float* b2 = (const float*)d_in[18];
  _Float16* ws = (_Float16*)d_ws;
  float* out = (float*)d_out;

  prep_weights<<<(WS_HALVES + 255) / 256, 256, 0, stream>>>(W_in, W_out, Wr, Wz, Wh, W1, W2, ws);

  // dynamic LDS: 65536+32768+16384+5*8192+4096+256 = 160000 B (< 320 KB/WGP)
  const size_t shmem = 160000;
  ggnn_kernel<<<4096, 256, shmem, stream>>>(prop_state, annotation, Aadj,
                                            br, bz, bh, b1, b2, ws, out);
}
